// AutoEncoder_88167088652779
// MI455X (gfx1250) — matrix operand order repaired and checked by reading, not compiled, not run on hardware
//
#include <hip/hip_runtime.h>
#include <math.h>

// ---------------- model constants (t5-base, from reference) ----------------
#define L_    12
#define H_    12
#define DK_   64
#define DM_   768
#define DFF_  3072
#define V_    32128
#define B_    2
#define S_    512
#define T_    512
#define VIS_  128
#define NEG_  (-1e9f)

typedef __attribute__((ext_vector_type(16))) __bf16 v16bf;
typedef __attribute__((ext_vector_type(8)))  float  v8f;

__device__ __forceinline__ unsigned short f2bf(float f) {
  __bf16 b = (__bf16)f;              // native v_cvt_pk_bf16_f32, RNE
  return __builtin_bit_cast(unsigned short, b);
}

// ---------------------------------------------------------------------------
// Block-tiled strided-batched GEMM:  C[bh] = beta*C[bh] + act(A[bh] x op(B[bh]))
// 256 threads = 8 waves. Block tile 256(M) x 64(N), K-step 32.
// A panel (256x32) and B panel (32x64, stored K-major) staged in LDS as bf16,
// converted once cooperatively (v_cvt_pk_bf16_f32). Staging issues ALL global
// loads first (one clause, one wait) then converts/stores. Each wave owns a
// 32x64 strip: 2 A + 4 B fragments per K-step, all ds_load_b128s issued before
// an 8x v_wmma_f32_16x16x32_bf16 burst (independent accumulators, partial
// s_wait_dscnt -> back-to-back issue on the matrix pipe).
// Model dims: M % 256 == 0, N % 64 == 0, K % 32 == 0 for every call (checked:
// M in {512,1024}, N in {64,512,768,3072,32128}, K in {64,512,768,3072}).
// ---------------------------------------------------------------------------
#define BM 256
#define BN 64
#define BK 32
#define LDK (BK + 8)   // row stride in ushorts: 80 bytes, keeps b128 reads aligned

struct GemmP {
  const float* A; const float* B; float* C;
  int M, N, K;
  int lda, ldb, ldc;
  long aO, aI, bO, bI, cO, cI;   // outer/inner batch strides (elements)
  int Hb;                        // inner batch count
  int transB;                    // 0: B is KxN, 1: B is NxK (use B^T)
  int beta;                      // 0: overwrite, 1: accumulate (residual)
  int relu;                      // fused relu on the product term
};

__global__ __launch_bounds__(256) void k_gemm(GemmP p) {
  __shared__ unsigned short As[BM][LDK];   // [m][k]  bf16
  __shared__ unsigned short Bs[BN][LDK];   // [n][k]  bf16 (K-major)

  const int tid  = threadIdx.x;
  const int lane = tid & 31;
  const int w    = tid >> 5;               // wave 0..7 -> 32-row M strip
  const int n0   = blockIdx.x * BN;
  const int m0   = blockIdx.y * BM;
  const int bh   = blockIdx.z;
  const int bo = bh / p.Hb, bi = bh % p.Hb;
  const float* A  = p.A + (long)bo * p.aO + (long)bi * p.aI;
  const float* Bm = p.B + (long)bo * p.bO + (long)bi * p.bI;
  float*       C  = p.C + (long)bo * p.cO + (long)bi * p.cI;

  const int  l15 = lane & 15;
  const bool hi  = lane >= 16;

  // Per-thread staging coordinates (loop-invariant).
  const int arow0 = (tid >> 3);            // A: rows tid/8 + 32*r
  const int akq   = (tid & 7) << 2;        // A: k quad
  const int bnrow = (tid >> 3);            // B(transB): n row (+32 for r=1)
  const int bkq   = (tid & 7) << 2;
  const int bkrow = (tid >> 4);            // B(normal): k row (+16 for r=1)
  const int bnq   = (tid & 15) << 2;

  v8f acc[2][4];
  #pragma unroll
  for (int i = 0; i < 2; ++i)
    #pragma unroll
    for (int t = 0; t < 4; ++t)
      acc[i][t] = (v8f){0.f,0.f,0.f,0.f,0.f,0.f,0.f,0.f};

  for (int k0 = 0; k0 < p.K; k0 += BK) {
    // ---- phase 1: issue ALL global loads (clause-friendly), then convert.
    float4 fa[8], fb[2];
    #pragma unroll
    for (int r = 0; r < 8; ++r) {
      const float* ap = A + (long)(m0 + arow0 + r * 32) * p.lda + k0 + akq;
      fa[r] = *(const float4*)ap;
    }
    if (p.transB) {                        // B is NxK: K contiguous in memory
      #pragma unroll
      for (int r = 0; r < 2; ++r)
        fb[r] = *(const float4*)(Bm + (long)(n0 + bnrow + r * 32) * p.ldb + k0 + bkq);
    } else {                               // B is KxN
      #pragma unroll
      for (int r = 0; r < 2; ++r)
        fb[r] = *(const float4*)(Bm + (long)(k0 + bkrow + r * 16) * p.ldb + n0 + bnq);
    }
    __builtin_prefetch(A + (long)(m0 + arow0) * p.lda + k0 + akq + BK, 0, 1);

    // ---- phase 2: cvt to bf16 + LDS stores
    #pragma unroll
    for (int r = 0; r < 8; ++r) {
      unsigned short* as = &As[arow0 + r * 32][akq];
      as[0] = f2bf(fa[r].x);
      as[1] = f2bf(fa[r].y);
      as[2] = f2bf(fa[r].z);
      as[3] = f2bf(fa[r].w);
    }
    if (p.transB) {
      #pragma unroll
      for (int r = 0; r < 2; ++r) {
        unsigned short* bs = &Bs[bnrow + r * 32][bkq];
        bs[0] = f2bf(fb[r].x);
        bs[1] = f2bf(fb[r].y);
        bs[2] = f2bf(fb[r].z);
        bs[3] = f2bf(fb[r].w);
      }
    } else {                               // transpose while staging
      #pragma unroll
      for (int r = 0; r < 2; ++r) {
        int kr = bkrow + r * 16;
        Bs[bnq + 0][kr] = f2bf(fb[r].x);
        Bs[bnq + 1][kr] = f2bf(fb[r].y);
        Bs[bnq + 2][kr] = f2bf(fb[r].z);
        Bs[bnq + 3][kr] = f2bf(fb[r].w);
      }
    }
    __syncthreads();

    // ---- load ALL fragments first (12x ds_load_b128), then WMMA burst.
    // A fragment layout (ISA 7.12.2, 16-bit A 16x32): lane<16 -> K{0..7,16..23},
    // lane>=16 -> K{8..15,24..31}. B uses the same K order, N = l15.
    union F { v16bf v; uint4 q[2]; };
    F af[2], bf[4];
    #pragma unroll
    for (int i = 0; i < 2; ++i) {
      const unsigned short* arow = &As[w * 32 + i * 16 + l15][0];
      af[i].q[0] = *(const uint4*)(arow + (hi ? 8 : 0));
      af[i].q[1] = *(const uint4*)(arow + 16 + (hi ? 8 : 0));
    }
    #pragma unroll
    for (int t = 0; t < 4; ++t) {
      const unsigned short* brow = &Bs[t * 16 + l15][0] + (hi ? 8 : 0);
      bf[t].q[0] = *(const uint4*)(brow);
      bf[t].q[1] = *(const uint4*)(brow + 16);
    }
    #pragma unroll
    for (int i = 0; i < 2; ++i)
      #pragma unroll
      for (int t = 0; t < 4; ++t)
        acc[i][t] = __builtin_amdgcn_wmma_f32_16x16x32_bf16(
            /*neg_a=*/false, af[i].v, /*neg_b=*/false, bf[t].v,
            /*c_mod=*/(short)0, acc[i][t], /*reuse_a=*/false, /*reuse_b=*/false);
    __syncthreads();
  }

  // ---- epilogue: C/D layout VGPR r -> M = r + (hi?8:0), N = l15
  #pragma unroll
  for (int i = 0; i < 2; ++i) {
    #pragma unroll
    for (int t = 0; t < 4; ++t) {
      #pragma unroll
      for (int r = 0; r < 8; ++r) {
        int m = m0 + w * 32 + i * 16 + r + (hi ? 8 : 0);
        int n = n0 + t * 16 + l15;
        float vv = acc[i][t][r];
        if (p.relu) vv = fmaxf(vv, 0.f);
        float* cp = C + (long)m * p.ldc + n;
        if (p.beta) vv += *cp;
        *cp = vv;
      }
    }
  }
}

// ------------------------- elementwise / reduction kernels ------------------
__global__ __launch_bounds__(256) void k_embed(const float* __restrict__ emb,
                                               const int* __restrict__ tok,
                                               float* __restrict__ out, int n) {
  int i = blockIdx.x * 256 + threadIdx.x;
  if (i >= n) return;
  int t = tok[i / DM_];
  out[i] = emb[(long)t * DM_ + (i % DM_)];
}

__global__ __launch_bounds__(256) void k_zero0(float* x, int rows) {
  int i = blockIdx.x * 256 + threadIdx.x;
  if (i < rows) x[(long)i * DM_] = 0.f;
}

// T5 RMS layernorm (no mean-sub, no bias) with optional output scale.
__global__ __launch_bounds__(256) void k_rmsnorm(const float* __restrict__ x,
                                                 const float* __restrict__ w,
                                                 float* __restrict__ out,
                                                 float scale) {
  int row = blockIdx.x;
  const float* xr = x + (long)row * DM_;
  float ss = 0.f;
  for (int i = threadIdx.x; i < DM_; i += 256) { float v = xr[i]; ss += v * v; }
  __shared__ float red[256];
  red[threadIdx.x] = ss;
  __syncthreads();
  for (int s = 128; s > 0; s >>= 1) {
    if (threadIdx.x < s) red[threadIdx.x] += red[threadIdx.x + s];
    __syncthreads();
  }
  float inv = rsqrtf(red[0] / (float)DM_ + 1e-6f) * scale;
  for (int i = threadIdx.x; i < DM_; i += 256)
    out[(long)row * DM_ + i] = xr[i] * inv * w[i];
}

__device__ __forceinline__ int bucket_bi(int rp) {  // bidirectional, 32 buckets
  int n = -rp;
  int ret = (n < 0) ? 16 : 0;
  n = n < 0 ? -n : n;
  float nf = (float)(n < 1 ? 1 : n);
  int large = 8 + (int)(logf(nf / 8.f) / logf(128.f / 8.f) * 8.f);
  if (large > 15) large = 15;
  return ret + (n < 8 ? n : large);
}

__device__ __forceinline__ int bucket_uni(int rp) {  // causal, 32 buckets
  int n = -rp;
  if (n < 0) n = 0;
  float nf = (float)(n < 1 ? 1 : n);
  int large = 16 + (int)(logf(nf / 16.f) / logf(128.f / 16.f) * 16.f);
  if (large > 31) large = 31;
  return n < 16 ? n : large;
}

__global__ __launch_bounds__(256) void k_enc_bias(const float* __restrict__ relb,
                                                  const int* __restrict__ slen,
                                                  float* __restrict__ bias) {
  int idx = blockIdx.x * 256 + threadIdx.x;
  if (idx >= B_ * H_ * S_ * S_) return;
  int k = idx % S_;
  int q = (idx / S_) % S_;
  int h = (idx / (S_ * S_)) % H_;
  int b = idx / (H_ * S_ * S_);
  float v = relb[bucket_bi(k - q) * H_ + h];
  v += (k < slen[b]) ? 0.f : NEG_;
  bias[idx] = v;
}

__global__ __launch_bounds__(256) void k_bias_upd(float* __restrict__ bias,
                                                  const float* __restrict__ pm) {
  int idx = blockIdx.x * 256 + threadIdx.x;
  if (idx >= B_ * H_ * S_ * S_) return;
  int k = idx % S_;
  int q = (idx / S_) % S_;
  int b = idx / (H_ * S_ * S_);
  if (q == k) bias[idx] = 0.f;                              // diagonal reset
  else        bias[idx] += pm[b * S_ + q] + pm[b * S_ + k];
}

__global__ __launch_bounds__(256) void k_prune(const float* __restrict__ x,
                                               const float* __restrict__ gumbel,
                                               const int* __restrict__ slen,
                                               const int* __restrict__ keep,
                                               float* __restrict__ pmprev,
                                               float* __restrict__ pm_out,
                                               float* __restrict__ pp_out,
                                               int layer) {
  int i = blockIdx.x * 256 + threadIdx.x;
  if (i >= B_ * S_) return;
  int b = i / S_, s = i % S_;
  float gate = x[(long)i * DM_] + gumbel[((long)b * L_ + layer) * S_ + s];
  float t = (gate + 3.0f) / 0.5f;
  // log_sigmoid, numerically stable
  float pm = (t >= 0.f) ? -log1pf(expf(-t)) : (t - log1pf(expf(t)));
  if (layer > 0) pm += pmprev[i];
  if (s == keep[b]) pm = 0.f;                               // forced keep
  pmprev[i] = pm;
  pm_out[((long)b * L_ + layer) * S_ + s] = pm;
  float pad = (s < slen[b]) ? 1.f : 0.f;
  pp_out[((long)b * L_ + layer) * S_ + s] = expf(pm / 8.0f) * pad;
}

__global__ __launch_bounds__(256) void k_dec_bias(const float* __restrict__ relb,
                                                  const int* __restrict__ slen,
                                                  const int* __restrict__ tlen,
                                                  float* __restrict__ bias) {
  int idx = blockIdx.x * 256 + threadIdx.x;
  if (idx >= B_ * H_ * T_ * T_) return;
  int k = idx % T_;
  int q = (idx / T_) % T_;
  int h = (idx / (T_ * T_)) % H_;
  int b = idx / (H_ * T_ * T_);
  float v = relb[bucket_uni(k - q) * H_ + h];
  float tm = ((k < slen[b]) ? 0.f : NEG_) +                 // ref uses src_lengths
             ((k <= q && k >= q - VIS_) ? 0.f : NEG_);      // local window
  if (k == 0 && q >= tlen[b]) tm = 0.f;                     // unblock col0 past len
  bias[idx] = v + tm;
}

__global__ __launch_bounds__(256) void k_cross_bias(const int* __restrict__ slen,
                                                    const float* __restrict__ pms,
                                                    float* __restrict__ cb) {
  int i = blockIdx.x * 256 + threadIdx.x;
  if (i >= B_ * S_) return;
  int b = i / S_, s = i % S_;
  cb[i] = ((s < slen[b]) ? 0.f : NEG_) + pms[((long)b * L_ + (L_ - 1)) * S_ + s];
}

// rows = B*H*Q, each block handles one row of length Kd; adds bias in-flight.
__global__ __launch_bounds__(256) void k_softmax(float* __restrict__ sc,
                                                 const float* __restrict__ bias,
                                                 int Kd, long sb_b, long sb_h,
                                                 long sb_q, int Hq, int Q) {
  int row = blockIdx.x;
  int q = row % Q;
  int h = (row / Q) % Hq;
  int b = row / (Q * Hq);
  float* r = sc + (long)row * Kd;
  const float* bi = bias + (long)b * sb_b + (long)h * sb_h + (long)q * sb_q;
  __shared__ float red[256];
  float mx = -3.4e38f;
  for (int i = threadIdx.x; i < Kd; i += 256) {
    float v = r[i] + bi[i];
    r[i] = v;
    mx = fmaxf(mx, v);
  }
  red[threadIdx.x] = mx;
  __syncthreads();
  for (int s = 128; s > 0; s >>= 1) {
    if (threadIdx.x < s) red[threadIdx.x] = fmaxf(red[threadIdx.x], red[threadIdx.x + s]);
    __syncthreads();
  }
  mx = red[0];
  __syncthreads();
  float sum = 0.f;
  for (int i = threadIdx.x; i < Kd; i += 256) {
    float e = expf(r[i] - mx);
    r[i] = e;
    sum += e;
  }
  red[threadIdx.x] = sum;
  __syncthreads();
  for (int s = 128; s > 0; s >>= 1) {
    if (threadIdx.x < s) red[threadIdx.x] += red[threadIdx.x + s];
    __syncthreads();
  }
  float inv = 1.f / red[0];
  for (int i = threadIdx.x; i < Kd; i += 256) r[i] *= inv;
}

// ------------------------------ host helpers -------------------------------
static inline void gemm(hipStream_t st, const float* A, const float* B, float* C,
                        int M, int N, int K, int lda, int ldb, int ldc,
                        long aO, long aI, long bO, long bI, long cO, long cI,
                        int Hb, int batches, int transB, int beta, int relu) {
  GemmP p{A, B, C, M, N, K, lda, ldb, ldc, aO, aI, bO, bI, cO, cI,
          Hb, transB, beta, relu};
  dim3 g(N / BN, M / BM, batches);
  k_gemm<<<g, dim3(256), 0, st>>>(p);
}

extern "C" void kernel_launch(void* const* d_in, const int* in_sizes, int n_in,
                              void* d_out, int out_size, void* d_ws, size_t ws_size,
                              hipStream_t stream) {
  (void)in_sizes; (void)n_in; (void)out_size; (void)ws_size;
  const float* embed   = (const float*)d_in[0];
  const float* enc_ln  = (const float*)d_in[1];
  const float* enc_aw  = (const float*)d_in[2];
  const float* enc_wi  = (const float*)d_in[3];
  const float* enc_wo  = (const float*)d_in[4];
  const float* enc_rb  = (const float*)d_in[5];
  const float* enc_fln = (const float*)d_in[6];
  const float* dec_ln  = (const float*)d_in[7];
  const float* dec_sw  = (const float*)d_in[8];
  const float* dec_cw  = (const float*)d_in[9];
  const float* dec_wi  = (const float*)d_in[10];
  const float* dec_wo  = (const float*)d_in[11];
  const float* dec_rb  = (const float*)d_in[12];
  const float* dec_fln = (const float*)d_in[13];
  const float* gumbel  = (const float*)d_in[14];
  const int* src_tok = (const int*)d_in[15];
  const int* src_len = (const int*)d_in[16];
  const int* tgt_tok = (const int*)d_in[17];
  const int* tgt_len = (const int*)d_in[18];
  const int* keep    = (const int*)d_in[19];

  // outputs (concatenated flat, reference return order)
  float* out    = (float*)d_out;
  float* mem    = out;                       // [B,S,DM]
  float* pms    = mem + (long)B_ * S_ * DM_; // [B,L,S]
  float* pps    = pms + (long)B_ * L_ * S_;  // [B,L,S]
  float* logits = pps + (long)B_ * L_ * S_;  // [B,T,V]

  // workspace layout
  const long NX  = (long)B_ * S_ * DM_;          // 786432
  const long NSC = (long)B_ * H_ * S_ * S_;      // 6291456
  float* ws     = (float*)d_ws;
  float* x      = ws;
  float* h      = x + NX;
  float* qb     = h + NX;
  float* kb     = qb + NX;
  float* vb     = kb + NX;
  float* ob     = vb + NX;
  float* ffn    = ob + NX;
  float* sc     = ffn + (long)B_ * S_ * DFF_;
  float* bias   = sc + NSC;                      // enc bias, reused as dec self bias
  float* pmprev = bias + NSC;
  float* crossb = pmprev + B_ * S_;

  const int TPB = 256;
  const int RS = B_ * S_;                        // 1024 token rows
  const int WDM = DM_ * DM_;                     // per-matrix stride in attn_w
  const int nbias = B_ * H_ * S_ * S_;
  const long sBh = (long)S_ * DM_;               // per-b stride in [B,S,768]
  const long sHQ = (long)H_ * S_ * S_;           // per-b stride in [B,H,S,S]
  const long sQQ = (long)S_ * S_;                // per-h stride

  // ================================ encoder ================================
  k_embed<<<(NX + TPB - 1) / TPB, TPB, 0, stream>>>(embed, src_tok, x, (int)NX);
  k_enc_bias<<<(nbias + TPB - 1) / TPB, TPB, 0, stream>>>(enc_rb, src_len, bias);

  for (int i = 0; i < L_; ++i) {
    const float* w = enc_aw + (long)i * 4 * WDM;
    k_zero0<<<(RS + TPB - 1) / TPB, TPB, 0, stream>>>(x, RS);
    k_rmsnorm<<<RS, TPB, 0, stream>>>(x, enc_ln + (i * 2 + 0) * DM_, h, 1.f);
    // q,k,v projections: [1024,768] x [768,768]
    gemm(stream, h, w + 0L * WDM, qb, RS, DM_, DM_, DM_, DM_, DM_, 0,0,0,0,0,0, 1, 1, 0, 0, 0);
    gemm(stream, h, w + 1L * WDM, kb, RS, DM_, DM_, DM_, DM_, DM_, 0,0,0,0,0,0, 1, 1, 0, 0, 0);
    gemm(stream, h, w + 2L * WDM, vb, RS, DM_, DM_, DM_, DM_, DM_, 0,0,0,0,0,0, 1, 1, 0, 0, 0);
    // scores[b,h] = Q[b,:,h*64:](512x64) x K^T  -> [B,H,S,S]
    gemm(stream, qb, kb, sc, S_, S_, DK_, DM_, DM_, S_,
         sBh, DK_, sBh, DK_, sHQ, sQQ, H_, B_ * H_, 1, 0, 0);
    k_softmax<<<B_ * H_ * S_, TPB, 0, stream>>>(sc, bias, S_, sHQ, sQQ, (long)S_, H_, S_);
    // O[b,h] = P(512x512) x V(512x64)
    gemm(stream, sc, vb, ob, S_, DK_, S_, S_, DM_, DM_,
         sHQ, sQQ, sBh, DK_, sBh, DK_, H_, B_ * H_, 0, 0, 0);
    // x += O x Wo (residual fused via beta)
    gemm(stream, ob, w + 3L * WDM, x, RS, DM_, DM_, DM_, DM_, DM_, 0,0,0,0,0,0, 1, 1, 0, 1, 0);
    // FFN
    k_rmsnorm<<<RS, TPB, 0, stream>>>(x, enc_ln + (i * 2 + 1) * DM_, h, 1.f);
    gemm(stream, h, enc_wi + (long)i * DM_ * DFF_, ffn, RS, DFF_, DM_,
         DM_, DFF_, DFF_, 0,0,0,0,0,0, 1, 1, 0, 0, 1);
    gemm(stream, ffn, enc_wo + (long)i * DFF_ * DM_, x, RS, DM_, DFF_,
         DFF_, DM_, DM_, 0,0,0,0,0,0, 1, 1, 0, 1, 0);
    // prune gate + bias update for next layer
    k_prune<<<(RS + TPB - 1) / TPB, TPB, 0, stream>>>(x, gumbel, src_len, keep,
                                                      pmprev, pms, pps, i);
    if (i < L_ - 1)
      k_bias_upd<<<(nbias + TPB - 1) / TPB, TPB, 0, stream>>>(bias, pmprev);
  }
  k_rmsnorm<<<RS, TPB, 0, stream>>>(x, enc_fln, mem, 1.f);

  // ================================ decoder ================================
  k_cross_bias<<<(RS + TPB - 1) / TPB, TPB, 0, stream>>>(src_len, pms, crossb);
  k_dec_bias<<<(nbias + TPB - 1) / TPB, TPB, 0, stream>>>(dec_rb, src_len, tgt_len, bias);
  k_embed<<<(NX + TPB - 1) / TPB, TPB, 0, stream>>>(embed, tgt_tok, x, (int)NX);

  for (int i = 0; i < L_; ++i) {
    const float* sw = dec_sw + (long)i * 4 * WDM;
    const float* cw = dec_cw + (long)i * 4 * WDM;
    // --- self attention ---
    k_rmsnorm<<<RS, TPB, 0, stream>>>(x, dec_ln + (i * 3 + 0) * DM_, h, 1.f);
    gemm(stream, h, sw + 0L * WDM, qb, RS, DM_, DM_, DM_, DM_, DM_, 0,0,0,0,0,0, 1, 1, 0, 0, 0);
    gemm(stream, h, sw + 1L * WDM, kb, RS, DM_, DM_, DM_, DM_, DM_, 0,0,0,0,0,0, 1, 1, 0, 0, 0);
    gemm(stream, h, sw + 2L * WDM, vb, RS, DM_, DM_, DM_, DM_, DM_, 0,0,0,0,0,0, 1, 1, 0, 0, 0);
    gemm(stream, qb, kb, sc, T_, T_, DK_, DM_, DM_, T_,
         sBh, DK_, sBh, DK_, sHQ, sQQ, H_, B_ * H_, 1, 0, 0);
    k_softmax<<<B_ * H_ * T_, TPB, 0, stream>>>(sc, bias, T_, sHQ, sQQ, (long)T_, H_, T_);
    gemm(stream, sc, vb, ob, T_, DK_, T_, T_, DM_, DM_,
         sHQ, sQQ, sBh, DK_, sBh, DK_, H_, B_ * H_, 0, 0, 0);
    gemm(stream, ob, sw + 3L * WDM, x, RS, DM_, DM_, DM_, DM_, DM_, 0,0,0,0,0,0, 1, 1, 0, 1, 0);
    // --- cross attention (KV from encoder mem) ---
    k_rmsnorm<<<RS, TPB, 0, stream>>>(x, dec_ln + (i * 3 + 1) * DM_, h, 1.f);
    gemm(stream, h,   cw + 0L * WDM, qb, RS, DM_, DM_, DM_, DM_, DM_, 0,0,0,0,0,0, 1, 1, 0, 0, 0);
    gemm(stream, mem, cw + 1L * WDM, kb, RS, DM_, DM_, DM_, DM_, DM_, 0,0,0,0,0,0, 1, 1, 0, 0, 0);
    gemm(stream, mem, cw + 2L * WDM, vb, RS, DM_, DM_, DM_, DM_, DM_, 0,0,0,0,0,0, 1, 1, 0, 0, 0);
    gemm(stream, qb, kb, sc, T_, S_, DK_, DM_, DM_, S_,
         sBh, DK_, sBh, DK_, sHQ, sQQ, H_, B_ * H_, 1, 0, 0);
    k_softmax<<<B_ * H_ * T_, TPB, 0, stream>>>(sc, crossb, S_, (long)S_, 0, 0, H_, T_);
    gemm(stream, sc, vb, ob, T_, DK_, S_, S_, DM_, DM_,
         sHQ, sQQ, sBh, DK_, sBh, DK_, H_, B_ * H_, 0, 0, 0);
    gemm(stream, ob, cw + 3L * WDM, x, RS, DM_, DM_, DM_, DM_, DM_, 0,0,0,0,0,0, 1, 1, 0, 1, 0);
    // --- FFN ---
    k_rmsnorm<<<RS, TPB, 0, stream>>>(x, dec_ln + (i * 3 + 2) * DM_, h, 1.f);
    gemm(stream, h, dec_wi + (long)i * DM_ * DFF_, ffn, RS, DFF_, DM_,
         DM_, DFF_, DFF_, 0,0,0,0,0,0, 1, 1, 0, 0, 1);
    gemm(stream, ffn, dec_wo + (long)i * DFF_ * DM_, x, RS, DM_, DFF_,
         DFF_, DM_, DM_, 0,0,0,0,0,0, 1, 1, 0, 1, 0);
  }

  // final LN with tied-embedding scale, then logits = y x embed^T
  k_rmsnorm<<<RS, TPB, 0, stream>>>(x, dec_fln, h, 0.03608439182435161f /* 768^-0.5 */);
  gemm(stream, h, embed, logits, RS, V_, DM_, DM_, DM_, V_,
       0,0,0,0,0,0, 1, 1, 1, 0, 0);
}
